// MixtureOfExperts_88665304859114
// MI455X (gfx1250) — compile-verified, matrix operand order checked by reading, not run on hardware
//
#include <hip/hip_runtime.h>

// Problem constants (match reference)
constexpr int N = 4096, D = 512, H = 1024, O = 512, E = 8;
constexpr int BT = 64;          // tokens per block in fused MLP kernel
constexpr int HPAD = H + 8;     // LDS row stride (bf16 elems): 2064B rows -> conflict-free b128 reads

typedef __bf16 bf16_t;
typedef __attribute__((ext_vector_type(16))) __bf16 v16bf;
typedef __attribute__((ext_vector_type(8)))  __bf16 v8bf;
typedef __attribute__((ext_vector_type(8)))  float  v8f;

#define WMMA_BF16(a, b, c) \
  __builtin_amdgcn_wmma_f32_16x16x32_bf16(false, (a), false, (b), (short)0, (c), false, false)

// Load one 16x32 bf16 fragment (A layout; B uses the same layout with pre-transposed
// weights so N-columns are contiguous rows). Lane L<16 holds row L, K = {0..7,16..23};
// lane L+16 holds row L, K = {8..15,24..31}. Two 16-byte loads per lane.
template <typename PT>
static __device__ __forceinline__ v16bf load_frag(PT p, int ld, int lane) {
  const int row  = lane & 15;
  const int half = lane >> 4;
  p += row * ld + half * 8;
  union { v16bf v; v8bf h[2]; } u;
  u.h[0] = *(const v8bf*)(p);
  u.h[1] = *(const v8bf*)(p + 16);
  return u.v;
}

// ---------------- prep kernels ----------------

__global__ void moe_cvt_bf16(const float* __restrict__ src, bf16_t* __restrict__ dst, int n) {
  int i = blockIdx.x * blockDim.x + threadIdx.x;
  if (i < n) dst[i] = (bf16_t)src[i];
}

// src: [E][R][C] fp32 row-major  ->  dst: [E][C][R] bf16 (transposed per expert)
__global__ __launch_bounds__(256) void moe_transpose_bf16(const float* __restrict__ src,
                                                          bf16_t* __restrict__ dst,
                                                          int R, int C) {
  __shared__ float tile[32][33];
  const int e  = blockIdx.z;
  const int c0 = blockIdx.x * 32;
  const int r0 = blockIdx.y * 32;
  const float* s = src + (size_t)e * R * C;
  bf16_t*      d = dst + (size_t)e * R * C;
#pragma unroll
  for (int rr = 0; rr < 32; rr += 8)
    tile[threadIdx.y + rr][threadIdx.x] =
        s[(size_t)(r0 + threadIdx.y + rr) * C + (c0 + threadIdx.x)];
  __syncthreads();
#pragma unroll
  for (int rr = 0; rr < 32; rr += 8)
    d[(size_t)(c0 + threadIdx.y + rr) * R + (r0 + threadIdx.x)] =
        (bf16_t)tile[threadIdx.x][threadIdx.y + rr];
}

// ---------------- gating: logits -> softmax -> top2 -> renorm weights ----------------

__global__ void moe_gate_kernel(const float* __restrict__ x, const float* __restrict__ Wg,
                                const float* __restrict__ bg, float* __restrict__ gates,
                                float* __restrict__ topi, float* __restrict__ wful) {
  int n = blockIdx.x * blockDim.x + threadIdx.x;
  if (n >= N) return;
  float acc[E];
#pragma unroll
  for (int e = 0; e < E; ++e) acc[e] = bg[e];
  const float* xr = x + (size_t)n * D;
  for (int d = 0; d < D; ++d) {
    float xv = xr[d];
    const float* wr = Wg + d * E;
#pragma unroll
    for (int e = 0; e < E; ++e) acc[e] += xv * wr[e];
  }
  float m = acc[0];
#pragma unroll
  for (int e = 1; e < E; ++e) m = fmaxf(m, acc[e]);
  float g[E], s = 0.f;
#pragma unroll
  for (int e = 0; e < E; ++e) { g[e] = __expf(acc[e] - m); s += g[e]; }
  float inv = 1.f / s;
#pragma unroll
  for (int e = 0; e < E; ++e) { g[e] *= inv; gates[(size_t)n * E + e] = g[e]; }
  int i0 = 0;
#pragma unroll
  for (int e = 1; e < E; ++e) if (g[e] > g[i0]) i0 = e;
  int i1 = (i0 == 0) ? 1 : 0;
#pragma unroll
  for (int e = 0; e < E; ++e) if (e != i0 && g[e] > g[i1]) i1 = e;
  float s2 = g[i0] + g[i1];
  float w0 = g[i0] / s2, w1 = g[i1] / s2;
#pragma unroll
  for (int e = 0; e < E; ++e)
    wful[(size_t)n * E + e] = (e == i0) ? w0 : ((e == i1) ? w1 : 0.f);
  topi[(size_t)n * 2 + 0] = (float)i0;
  topi[(size_t)n * 2 + 1] = (float)i1;
}

// ---------------- fused per-expert MLP: out += wful[:,e] * (relu(x W1 + b1) W2 + b2) ----------------

__global__ __launch_bounds__(256) void moe_mlp_kernel(
    const bf16_t* __restrict__ xb,   // [N][D] bf16
    const bf16_t* __restrict__ W1b,  // [E][H][D] bf16 (transposed)
    const float*  __restrict__ b1,   // [E][H]
    const bf16_t* __restrict__ W2b,  // [E][O][H] bf16 (transposed)
    const float*  __restrict__ b2,   // [E][O]
    const float*  __restrict__ wful, // [N][E] combine weights (0 if not selected)
    float* __restrict__ out,         // [N][O], accumulated across expert launches
    int e) {
  extern __shared__ bf16_t hs[];  // [BT][HPAD] bf16 hidden tile
  const int lane = threadIdx.x & 31;
  const int wv   = threadIdx.x >> 5;  // 8 waves
  const int tok0 = blockIdx.x * BT;
  const int half = lane >> 4;
  const int col  = lane & 15;

  // ---- phase 1: h = relu(x W1 + b1), 64 x 1024, staged into LDS ----
  for (int t = wv; t < (BT / 32) * (H / 32); t += 8) {
    const int mi = t & 1;        // 32-token sub-block
    const int ni = t >> 1;       // 32-wide h sub-block
    v8f acc[2][2] = {};
    const bf16_t* Ab = xb + (size_t)(tok0 + mi * 32) * D;
    const bf16_t* Bb = W1b + ((size_t)e * H + ni * 32) * D;
    for (int k0 = 0; k0 < D; k0 += 32) {
      v16bf a0 = load_frag(Ab + k0, D, lane);
      v16bf a1 = load_frag(Ab + 16 * D + k0, D, lane);
      v16bf b0 = load_frag(Bb + k0, D, lane);
      v16bf bb1 = load_frag(Bb + 16 * D + k0, D, lane);
      acc[0][0] = WMMA_BF16(a0, b0, acc[0][0]);
      acc[0][1] = WMMA_BF16(a0, bb1, acc[0][1]);
      acc[1][0] = WMMA_BF16(a1, b0, acc[1][0]);
      acc[1][1] = WMMA_BF16(a1, bb1, acc[1][1]);
    }
#pragma unroll
    for (int i = 0; i < 2; ++i)
#pragma unroll
      for (int j = 0; j < 2; ++j) {
        const int hcol = ni * 32 + j * 16 + col;
        const float bias = b1[e * H + hcol];
#pragma unroll
        for (int r = 0; r < 8; ++r) {
          const int tl = mi * 32 + i * 16 + half * 8 + r;
          float v = acc[i][j][r] + bias;
          hs[tl * HPAD + hcol] = (bf16_t)(v > 0.f ? v : 0.f);
        }
      }
  }
  __syncthreads();

  // ---- phase 2: y = h W2 + b2; out += w * y ----
  for (int t = wv; t < (BT / 32) * (O / 32); t += 8) {
    const int mi = t & 1;
    const int ni = t >> 1;
    v8f acc[2][2] = {};
    const bf16_t* Bb = W2b + ((size_t)e * O + ni * 32) * H;
    for (int k0 = 0; k0 < H; k0 += 32) {
      v16bf a0 = load_frag(hs + (mi * 32) * HPAD + k0, HPAD, lane);
      v16bf a1 = load_frag(hs + (mi * 32 + 16) * HPAD + k0, HPAD, lane);
      v16bf b0 = load_frag(Bb + k0, H, lane);
      v16bf bb1 = load_frag(Bb + 16 * H + k0, H, lane);
      acc[0][0] = WMMA_BF16(a0, b0, acc[0][0]);
      acc[0][1] = WMMA_BF16(a0, bb1, acc[0][1]);
      acc[1][0] = WMMA_BF16(a1, b0, acc[1][0]);
      acc[1][1] = WMMA_BF16(a1, bb1, acc[1][1]);
    }
#pragma unroll
    for (int i = 0; i < 2; ++i)
#pragma unroll
      for (int j = 0; j < 2; ++j) {
        const int ocol = ni * 32 + j * 16 + col;
        const float bias = b2[e * O + ocol];
#pragma unroll
        for (int r = 0; r < 8; ++r) {
          const int tok = tok0 + mi * 32 + i * 16 + half * 8 + r;
          const float w = wful[(size_t)tok * E + e];
          const float y = acc[i][j][r] + bias;
          float* op = out + (size_t)tok * O + ocol;
          *op = *op + w * y;  // experts launched sequentially -> deterministic RMW
        }
      }
  }
}

// ---------------- host entry ----------------

extern "C" void kernel_launch(void* const* d_in, const int* in_sizes, int n_in,
                              void* d_out, int out_size, void* d_ws, size_t ws_size,
                              hipStream_t stream) {
  const float* x  = (const float*)d_in[0];
  const float* W1 = (const float*)d_in[1];
  const float* b1 = (const float*)d_in[2];
  const float* W2 = (const float*)d_in[3];
  const float* b2 = (const float*)d_in[4];
  const float* Wg = (const float*)d_in[5];
  const float* bg = (const float*)d_in[6];
  // d_in[7] = k (always 2 for this problem)

  float* out   = (float*)d_out;                  // [N][O]
  float* gates = out + (size_t)N * O;            // [N][E]
  float* topi  = gates + (size_t)N * E;          // [N][2] (indices stored as floats)

  char* ws = (char*)d_ws;
  bf16_t* xb  = (bf16_t*)ws; ws += (size_t)N * D * sizeof(bf16_t);
  bf16_t* W1b = (bf16_t*)ws; ws += (size_t)E * D * H * sizeof(bf16_t);
  bf16_t* W2b = (bf16_t*)ws; ws += (size_t)E * H * O * sizeof(bf16_t);
  float*  wfl = (float*)ws;  ws += (size_t)N * E * sizeof(float);

  hipMemsetAsync(d_out, 0, (size_t)N * O * sizeof(float), stream);

  moe_cvt_bf16<<<(N * D + 255) / 256, 256, 0, stream>>>(x, xb, N * D);

  dim3 tb(32, 8);
  moe_transpose_bf16<<<dim3(H / 32, D / 32, E), tb, 0, stream>>>(W1, W1b, D, H);
  moe_transpose_bf16<<<dim3(O / 32, H / 32, E), tb, 0, stream>>>(W2, W2b, H, O);

  moe_gate_kernel<<<N / 256, 256, 0, stream>>>(x, Wg, bg, gates, topi, wfl);

  const size_t smem = (size_t)BT * HPAD * sizeof(bf16_t);  // 132096 B < 320 KB LDS/WGP
  hipFuncSetAttribute(reinterpret_cast<const void*>(moe_mlp_kernel),
                      hipFuncAttributeMaxDynamicSharedMemorySize, (int)smem);
  for (int e = 0; e < E; ++e) {
    moe_mlp_kernel<<<N / BT, 256, smem, stream>>>(xb, W1b, b1, W2b, b2, wfl, out, e);
  }
}
